// adarnn_38946763440818
// MI455X (gfx1250) — matrix operand order repaired and checked by reading, not compiled, not run on hardware
//
#include <hip/hip_runtime.h>
#include <stddef.h>

// ---------------- problem constants (match reference setup) ----------------
#define T_SEQ   120
#define BATCH   256
#define HID     512
#define G3      1536          // 3*HID
#define D_IN0   64
#define OUT_SEQ 24
#define HALF    128           // BATCH/2
#define GCOL    (T_SEQ*2*HID) // 122880
#define BT      (BATCH*T_SEQ) // 30720

typedef __attribute__((ext_vector_type(16))) __bf16 v16bf;
typedef __attribute__((ext_vector_type(8)))  float  v8f;

union BfFrag { v16bf v; uint4 q[2]; unsigned short s[16]; };

__device__ __forceinline__ unsigned short f2bf(float f) {
  union { float f; unsigned u; } x; x.f = f;
  unsigned u = x.u + 0x7FFFu + ((x.u >> 16) & 1u);   // round-to-nearest-even
  return (unsigned short)(u >> 16);
}
__device__ __forceinline__ float sigmoidf_(float x) { return 1.0f / (1.0f + expf(-x)); }

__device__ __forceinline__ v8f wmma_bf16(v16bf a, v16bf b, v8f c) {
  // D = A(16x32 bf16) * B(32x16 bf16) + C(16x16 f32)
  return __builtin_amdgcn_wmma_f32_16x16x32_bf16(false, a, false, b, (short)0, c,
                                                 false, false);
}

// A fragment: 16x32 bf16 tile from row-major source, base at (tile_row0, tile_k0).
// Lane m = L&15 holds row m; slots = row[kh..kh+7] and row[16+kh..16+kh+7]
// (kh = (L>>4)*8) -> two contiguous, 16B-aligned b128 loads per lane.
__device__ __forceinline__ v16bf load_a_bf16(const unsigned short* A, int lda, int lane) {
  int m  = lane & 15;
  int kh = (lane >> 4) << 3;
  const unsigned short* row = A + (size_t)m * lda + kh;
  BfFrag f;
  f.q[0] = *(const uint4*)(row);
  f.q[1] = *(const uint4*)(row + 16);
  return f.v;
}

// B fragment from pre-packed fragment-major weights:
// Wpk[((kt*NT + nt)*32 + lane)*16 + idx]  -> two b128 loads, lanes fully coalesced.
__device__ __forceinline__ v16bf load_b_packed(const unsigned short* Wpk, int NT,
                                               int kt, int nt, int lane) {
  const unsigned short* p = Wpk + (((size_t)kt * NT + nt) * 32 + lane) * 16;
  BfFrag f;
  f.q[0] = ((const uint4*)p)[0];
  f.q[1] = ((const uint4*)p)[1];
  return f.v;
}

// ---------------- prep kernels ----------------

__global__ void k_zero_f32(float* p, int n) {
  int i = blockIdx.x * blockDim.x + threadIdx.x;
  if (i < n) p[i] = 0.0f;
}

// Pack W (Nreal x K row-major fp32; B[k][n] = W[n][k]) into WMMA B-fragment order,
// padding N to Npad (multiple of 16) with zeros.
// Wpk[((kt*(Npad/16) + nt)*32 + lane)*16 + idx] = bf16(W[n*K + k]),
//   n = nt*16 + (lane&15),  k = kt*32 + (lane>>4)*8 + (idx&7) + (idx>>3)*16.
__global__ void k_pack_b(const float* W, unsigned short* Wpk, int Npad, int Nreal, int K) {
  size_t total = (size_t)Npad * K;
  size_t o = (size_t)blockIdx.x * blockDim.x + threadIdx.x;
  if (o >= total) return;
  int idx  = (int)(o & 15);
  int lane = (int)((o >> 4) & 31);
  size_t tile = o >> 9;
  int NT = Npad >> 4;
  int kt = (int)(tile / NT), nt = (int)(tile % NT);
  int n = nt * 16 + (lane & 15);
  int k = kt * 32 + ((lane >> 4) << 3) + (idx & 7) + ((idx >> 3) << 4);
  Wpk[o] = (n < Nreal) ? f2bf(W[(size_t)n * K + k]) : (unsigned short)0;
}

// x0[b][t][d] = concat(X_lag, X_cov)[t][b][d] as bf16
__global__ void k_build_x0(const float* Xlag, const float* Xcov, unsigned short* x0b) {
  int idx = blockIdx.x * blockDim.x + threadIdx.x;
  if (idx >= BT * D_IN0) return;
  int d = idx % D_IN0;
  int t = (idx / D_IN0) % T_SEQ;
  int b = idx / (D_IN0 * T_SEQ);
  float v = (d < 32) ? Xlag[((size_t)t * BATCH + b) * 32 + d]
                     : Xcov[((size_t)t * BATCH + b) * 32 + (d - 32)];
  x0b[idx] = f2bf(v);
}

// ---------------- WMMA GEMM: C(MxN,f32) = A(MxK,bf16) * Bpk + bias ----------------
__global__ void k_gemm_bias(const unsigned short* A, const unsigned short* Bpk,
                            const float* bias, float* C, int M, int N, int K) {
  int wave = (blockIdx.x * blockDim.x + threadIdx.x) >> 5;
  int lane = threadIdx.x & 31;
  int NT = N >> 4;
  int mt = wave / NT, nt = wave % NT;
  if (mt >= (M >> 4)) return;
  const unsigned short* Ab = A + (size_t)(mt * 16) * K;
  v8f c = {};
  for (int k0 = 0; k0 < K; k0 += 32) {
    v16bf a = load_a_bf16(Ab + k0, K, lane);
    v16bf b = load_b_packed(Bpk, NT, k0 >> 5, nt, lane);
    c = wmma_bf16(a, b, c);
  }
  int n = lane & 15, mh = (lane >> 4) << 3;
  float bv = bias[nt * 16 + n];
#pragma unroll
  for (int j = 0; j < 8; ++j)
    C[(size_t)(mt * 16 + mh + j) * N + nt * 16 + n] = c[j] + bv;
}

// ---------------- fused GRU step ----------------
// One wave computes the r/z/n 16x16 gh-tiles for one (batch-tile, hid-tile),
// sharing the A fragment, then applies gates in-register.
// 512 waves: 16 M-tiles x 32 N-tiles over HID. Whh pre-packed with N = G3.
__global__ void k_gru_step(const unsigned short* hprevb, int ldhb,
                           const float* hprevf, int ldhf,
                           const unsigned short* WhhPk,  // packed, NT = G3/16 = 96
                           const float* gx,              // [B][T][G3], includes b_ih
                           const float* bhh,             // G3
                           float* outf, unsigned short* outb, int t) {
  int wave = (blockIdx.x * blockDim.x + threadIdx.x) >> 5;
  int lane = threadIdx.x & 31;
  if (wave >= 16 * 32) return;
  int mt = wave >> 5;
  int nt = wave & 31;             // hid tile: columns c0..c0+15 of HID
  int r0 = mt * 16, c0 = nt * 16;
  const int NT = G3 >> 4;         // 96
  v8f cr = {}, cz = {}, cn = {};
  for (int k0 = 0; k0 < HID; k0 += 32) {
    int kt = k0 >> 5;
    v16bf a  = load_a_bf16(hprevb + (size_t)r0 * ldhb + k0, ldhb, lane);
    v16bf br = load_b_packed(WhhPk, NT, kt, nt,                    lane);
    v16bf bz = load_b_packed(WhhPk, NT, kt, nt + (HID >> 4),       lane);
    v16bf bn = load_b_packed(WhhPk, NT, kt, nt + 2 * (HID >> 4),   lane);
    cr = wmma_bf16(a, br, cr);
    cz = wmma_bf16(a, bz, cz);
    cn = wmma_bf16(a, bn, cn);
  }
  int n = lane & 15, mh = (lane >> 4) << 3;
  int c = c0 + n;
  float br_ = bhh[c], bz_ = bhh[HID + c], bn_ = bhh[2 * HID + c];
#pragma unroll
  for (int j = 0; j < 8; ++j) {
    int b = r0 + mh + j;
    const float* gxp = gx + ((size_t)b * T_SEQ + t) * G3;
    float ghr = cr[j] + br_;
    float ghz = cz[j] + bz_;
    float ghn = cn[j] + bn_;
    float r  = sigmoidf_(gxp[c] + ghr);
    float z  = sigmoidf_(gxp[HID + c] + ghz);
    float ng = tanhf(gxp[2 * HID + c] + r * ghn);
    float hold = hprevf[(size_t)b * ldhf + c];
    float hnew = (1.0f - z) * ng + z * hold;
    size_t oidx = ((size_t)b * T_SEQ + t) * HID + c;
    outf[oidx] = hnew;
    outb[oidx] = f2bf(hnew);
  }
}

// ---------------- gate GEMM: z(128x128) = x_all(128xGCOL) @ gate_Wl^T ----------------
// A gathered from bf16 GRU outputs with b128 loads (8-aligned col runs never cross
// the 512-col half boundary); B pre-packed bf16 fragments (N padded 120->128).
__global__ void k_gate_gemm(const unsigned short* outb, const unsigned short* gWpk,
                            float* zmat) {
  int wave = (blockIdx.x * blockDim.x + threadIdx.x) >> 5;
  int lane = threadIdx.x & 31;
  if (wave >= 64) return;
  int mt = wave >> 3, nt = wave & 7;
  int r0 = mt * 16, c0 = nt * 16;
  int m  = lane & 15;
  int kh = (lane >> 4) << 3;
  int i  = r0 + m;
  v8f acc = {};
  for (int k0 = 0; k0 < GCOL; k0 += 32) {
    BfFrag fa;
#pragma unroll
    for (int blk = 0; blk < 2; ++blk) {
      int col0 = k0 + kh + 16 * blk;       // multiple of 8
      int tt = col0 >> 10;                 // 2H = 1024 per timestep
      int cc = col0 & 1023;
      int ib = (cc >= HID) ? (i + HALF) : i;
      const unsigned short* p =
          outb + ((size_t)ib * T_SEQ + tt) * HID + (cc & (HID - 1));
      fa.q[blk] = *(const uint4*)p;
    }
    v16bf fb = load_b_packed(gWpk, 8, k0 >> 5, nt, lane);
    acc = wmma_bf16(fa.v, fb, acc);
  }
  int n = lane & 15, mh = (lane >> 4) << 3;
#pragma unroll
  for (int j = 0; j < 8; ++j)
    zmat[(size_t)(r0 + mh + j) * 128 + c0 + n] = acc[j];
}

// ---------------- BN over batch + sigmoid-mean + softmax -> gate weights ----------------
__global__ void k_gate_stats(const float* zmat, const float* gate_b,
                             const float* bn_g, const float* bn_b,
                             float* gw_out, float* gwbuf, int l) {
  __shared__ float sw[128];
  __shared__ float smax, ssum;
  int j = threadIdx.x;
  float wm = -1e30f;
  if (j < T_SEQ) {
    float zb = gate_b[l * T_SEQ + j];
    float mean = 0.0f;
    for (int i = 0; i < HALF; ++i) mean += zmat[i * 128 + j];
    mean = mean / HALF + zb;
    float var = 0.0f;
    for (int i = 0; i < HALF; ++i) {
      float d = zmat[i * 128 + j] + zb - mean;
      var += d * d;
    }
    var /= HALF;
    float g = bn_g[l * T_SEQ + j], be = bn_b[l * T_SEQ + j];
    float inv = rsqrtf(var + 1e-5f);
    float s = 0.0f;
    for (int i = 0; i < HALF; ++i) {
      float zn = g * (zmat[i * 128 + j] + zb - mean) * inv + be;
      s += 1.0f / (1.0f + expf(-zn));
    }
    wm = s / HALF;
  }
  sw[j] = wm;
  __syncthreads();
  if (j == 0) {
    float mx = -1e30f;
    for (int q = 0; q < T_SEQ; ++q) mx = fmaxf(mx, sw[q]);
    float su = 0.0f;
    for (int q = 0; q < T_SEQ; ++q) su += expf(sw[q] - mx);
    smax = mx; ssum = su;
  }
  __syncthreads();
  if (j < T_SEQ) {
    float w = expf(wm - smax) / ssum;
    gw_out[j] = w;
    gwbuf[l * T_SEQ + j] = w;
  }
}

// ---------------- transfer-loss helpers ----------------
__global__ void k_means(const float* outf, float* ms, float* mt) {
  int idx = blockIdx.x * blockDim.x + threadIdx.x;
  if (idx >= T_SEQ * HID) return;
  int t = idx / HID, h = idx % HID;
  float s0 = 0.0f, s1 = 0.0f;
  for (int i = 0; i < HALF; ++i) {
    s0 += outf[((size_t)i * T_SEQ + t) * HID + h];
    s1 += outf[((size_t)(i + HALF) * T_SEQ + t) * HID + h];
  }
  ms[idx] = s0 / HALF;
  mt[idx] = s1 / HALF;
}

__global__ void k_norms(const float* ms, const float* mt, float* ns, float* ntn) {
  int t = blockIdx.x * blockDim.x + threadIdx.x;
  if (t >= T_SEQ) return;
  float a = 0.0f, b = 0.0f;
  for (int h = 0; h < HID; ++h) {
    float x = ms[t * HID + h]; a += x * x;
    float y = mt[t * HID + h]; b += y * y;
  }
  ns[t] = a; ntn[t] = b;
}

__global__ void k_transfer(const float* ms, const float* mt, const float* ns,
                           const float* ntn, const float* gwbuf, int l,
                           const int* len_win_p, float* loss_acc) {
  __shared__ float red[256];
  __shared__ float accs;
  int t1 = blockIdx.x;
  int tid = threadIdx.x;
  int lw = len_win_p[0];
  if (tid == 0) accs = 0.0f;
  __syncthreads();
  for (int t2 = 0; t2 < T_SEQ; ++t2) {
    int ad = (t1 > t2) ? (t1 - t2) : (t2 - t1);
    if (ad > lw) continue;                       // uniform across block
    float p = 0.0f;
    for (int h = tid; h < HID; h += 256) p += ms[t1 * HID + h] * mt[t2 * HID + h];
    red[tid] = p; __syncthreads();
    for (int s = 128; s > 0; s >>= 1) {
      if (tid < s) red[tid] += red[tid + s];
      __syncthreads();
    }
    if (tid == 0) accs += ns[t1] + ntn[t2] - 2.0f * red[0];
    __syncthreads();
  }
  if (tid == 0) atomicAdd(loss_acc, gwbuf[l * T_SEQ + t1] * accs);
}

// ---------------- fc head + NLL ----------------
__global__ void k_fc_loss(const float* out1f, const float* fcW, const float* fcb,
                          const float* y, float* out_loc, float* out_scale,
                          float* lp_acc) {
  int idx = blockIdx.x * blockDim.x + threadIdx.x;
  if (idx >= OUT_SEQ * BATCH) return;
  int tt = idx / BATCH, b = idx % BATCH;
  int t = T_SEQ - OUT_SEQ + tt;
  float lin = fcb[0];
  const float* hp = out1f + ((size_t)b * T_SEQ + t) * HID;
  for (int h = 0; h < HID; ++h) lin += hp[h] * fcW[h];
  float loc = lin;
  float sp = (lin > 20.0f) ? lin : log1pf(expf(lin));
  float scale = sp + 1e-6f;
  out_loc[idx] = loc;
  out_scale[idx] = scale;
  float yt = y[(size_t)t * BATCH + b];
  float dd = (yt - loc) / scale;
  float lp = -0.5f * dd * dd - logf(scale) - 0.5f * logf(6.2831853071795864769f);
  atomicAdd(lp_acc, lp);
}

__global__ void k_finalize(const float* lp_acc, const float* lt_acc, float* out) {
  float ly = -lp_acc[0] / (float)(OUT_SEQ * BATCH);
  out[0] = ly + lt_acc[0];   // GAMMA = 1.0
  out[1] = ly;
}

// ---------------- host orchestration ----------------
static inline void* carve(char*& w, size_t bytes) {
  void* p = (void*)w;
  w += (bytes + 255) & ~(size_t)255;
  return p;
}

extern "C" void kernel_launch(void* const* d_in, const int* in_sizes, int n_in,
                              void* d_out, int out_size, void* d_ws, size_t ws_size,
                              hipStream_t stream) {
  (void)in_sizes; (void)n_in; (void)out_size; (void)ws_size;
  const float* X_cov = (const float*)d_in[1];
  const float* X_lag = (const float*)d_in[2];
  const float* y_in  = (const float*)d_in[3];
  const int*   lenw  = (const int*)d_in[5];
  const float* Wih0  = (const float*)d_in[6];
  const float* Whh0  = (const float*)d_in[7];
  const float* bih0  = (const float*)d_in[8];
  const float* bhh0  = (const float*)d_in[9];
  const float* Wih1  = (const float*)d_in[10];
  const float* Whh1  = (const float*)d_in[11];
  const float* bih1  = (const float*)d_in[12];
  const float* bhh1  = (const float*)d_in[13];
  const float* gateW = (const float*)d_in[14];
  const float* gateB = (const float*)d_in[15];
  const float* bnG   = (const float*)d_in[16];
  const float* bnB   = (const float*)d_in[17];
  const float* fcW   = (const float*)d_in[18];
  const float* fcb   = (const float*)d_in[19];
  float* out = (float*)d_out;

  // ---- workspace carve (~451 MB; gx reused across layers) ----
  char* w = (char*)d_ws;
  float*          acc     = (float*)         carve(w, 2 * sizeof(float));
  float*          gwbuf   = (float*)         carve(w, 2 * T_SEQ * sizeof(float));
  float*          zmat    = (float*)         carve(w, 128 * 128 * sizeof(float));
  float*          zslabF  = (float*)         carve(w, (size_t)BATCH * HID * 4);
  unsigned short* zslabB  = (unsigned short*)carve(w, (size_t)BATCH * HID * 2);
  unsigned short* Wih0pk  = (unsigned short*)carve(w, (size_t)D_IN0 * G3 * 2);
  unsigned short* Whh0pk  = (unsigned short*)carve(w, (size_t)HID * G3 * 2);
  unsigned short* Wih1pk  = (unsigned short*)carve(w, (size_t)HID * G3 * 2);
  unsigned short* Whh1pk  = (unsigned short*)carve(w, (size_t)HID * G3 * 2);
  unsigned short* gW0pk   = (unsigned short*)carve(w, (size_t)GCOL * 128 * 2);
  unsigned short* gW1pk   = (unsigned short*)carve(w, (size_t)GCOL * 128 * 2);
  unsigned short* x0b     = (unsigned short*)carve(w, (size_t)BT * D_IN0 * 2);
  float*          gx      = (float*)         carve(w, (size_t)BT * G3 * 4);
  float*          out0f   = (float*)         carve(w, (size_t)BT * HID * 4);
  unsigned short* out0b   = (unsigned short*)carve(w, (size_t)BT * HID * 2);
  float*          out1f   = (float*)         carve(w, (size_t)BT * HID * 4);
  unsigned short* out1b   = (unsigned short*)carve(w, (size_t)BT * HID * 2);
  float*          msb     = (float*)         carve(w, (size_t)T_SEQ * HID * 4);
  float*          mtb     = (float*)         carve(w, (size_t)T_SEQ * HID * 4);
  float*          nsb     = (float*)         carve(w, T_SEQ * sizeof(float));
  float*          ntb     = (float*)         carve(w, T_SEQ * sizeof(float));

  // ---- zero accumulators + zero h-slabs (deterministic per call) ----
  k_zero_f32<<<1, 256, 0, stream>>>(acc, 2);
  k_zero_f32<<<(BATCH * HID + 255) / 256, 256, 0, stream>>>(zslabF, BATCH * HID);
  k_zero_f32<<<(BATCH * HID / 2 + 255) / 256, 256, 0, stream>>>((float*)zslabB, BATCH * HID / 2);

  // ---- pack weights into bf16 WMMA B-fragment order ----
  k_pack_b<<<(G3 * D_IN0 + 255) / 256, 256, 0, stream>>>(Wih0, Wih0pk, G3, G3, D_IN0);
  k_pack_b<<<(G3 * HID + 255) / 256, 256, 0, stream>>>(Whh0, Whh0pk, G3, G3, HID);
  k_pack_b<<<(G3 * HID + 255) / 256, 256, 0, stream>>>(Wih1, Wih1pk, G3, G3, HID);
  k_pack_b<<<(G3 * HID + 255) / 256, 256, 0, stream>>>(Whh1, Whh1pk, G3, G3, HID);
  {
    int blocks = (int)(((size_t)128 * GCOL + 255) / 256);
    k_pack_b<<<blocks, 256, 0, stream>>>(gateW, gW0pk, 128, T_SEQ, GCOL);
    k_pack_b<<<blocks, 256, 0, stream>>>(gateW + (size_t)T_SEQ * GCOL, gW1pk, 128, T_SEQ, GCOL);
  }

  // ---- x0 = concat(X_lag, X_cov), batch-first, bf16 ----
  k_build_x0<<<(BT * D_IN0 + 255) / 256, 256, 0, stream>>>(X_lag, X_cov, x0b);

  // ---- layer 0: gx = x0 @ Wih0^T + b_ih0 ----
  {
    int waves = (BT / 16) * (G3 / 16);
    k_gemm_bias<<<waves / 4, 128, 0, stream>>>(x0b, Wih0pk, bih0, gx, BT, G3, D_IN0);
  }
  // ---- layer 0 recurrent scan ----
  for (int t = 0; t < T_SEQ; ++t) {
    const unsigned short* hb = (t == 0) ? zslabB : (out0b + (size_t)(t - 1) * HID);
    const float*          hf = (t == 0) ? zslabF : (out0f + (size_t)(t - 1) * HID);
    int ldh = (t == 0) ? HID : (T_SEQ * HID);
    k_gru_step<<<128, 128, 0, stream>>>(hb, ldh, hf, ldh, Whh0pk, gx, bhh0, out0f, out0b, t);
  }
  // ---- layer 0 gates ----
  k_gate_gemm<<<16, 128, 0, stream>>>(out0b, gW0pk, zmat);
  k_gate_stats<<<1, 128, 0, stream>>>(zmat, gateB, bnG, bnB, out + 12290, gwbuf, 0);

  // ---- layer 1: gx = out0 @ Wih1^T + b_ih1 (reuse gx buffer) ----
  {
    int waves = (BT / 16) * (G3 / 16);
    k_gemm_bias<<<waves / 4, 128, 0, stream>>>(out0b, Wih1pk, bih1, gx, BT, G3, HID);
  }
  for (int t = 0; t < T_SEQ; ++t) {
    const unsigned short* hb = (t == 0) ? zslabB : (out1b + (size_t)(t - 1) * HID);
    const float*          hf = (t == 0) ? zslabF : (out1f + (size_t)(t - 1) * HID);
    int ldh = (t == 0) ? HID : (T_SEQ * HID);
    k_gru_step<<<128, 128, 0, stream>>>(hb, ldh, hf, ldh, Whh1pk, gx, bhh1, out1f, out1b, t);
  }
  // ---- layer 1 gates ----
  k_gate_gemm<<<16, 128, 0, stream>>>(out1b, gW1pk, zmat);
  k_gate_stats<<<1, 128, 0, stream>>>(zmat, gateB, bnG, bnB, out + 12290 + T_SEQ, gwbuf, 1);

  // ---- transfer loss (layer 0 then layer 1, reusing mean buffers) ----
  k_means<<<(T_SEQ * HID + 255) / 256, 256, 0, stream>>>(out0f, msb, mtb);
  k_norms<<<1, 128, 0, stream>>>(msb, mtb, nsb, ntb);
  k_transfer<<<T_SEQ, 256, 0, stream>>>(msb, mtb, nsb, ntb, gwbuf, 0, lenw, acc + 1);

  k_means<<<(T_SEQ * HID + 255) / 256, 256, 0, stream>>>(out1f, msb, mtb);
  k_norms<<<1, 128, 0, stream>>>(msb, mtb, nsb, ntb);
  k_transfer<<<T_SEQ, 256, 0, stream>>>(msb, mtb, nsb, ntb, gwbuf, 1, lenw, acc + 1);

  // ---- fc head + NLL ----
  k_fc_loss<<<(OUT_SEQ * BATCH + 255) / 256, 256, 0, stream>>>(
      out1f, fcW, fcb, y_in, out + 2, out + 2 + OUT_SEQ * BATCH, acc);

  // ---- totals ----
  k_finalize<<<1, 1, 0, stream>>>(acc, acc + 1, out);
}
